// TransformerBlock_60705067761802
// MI455X (gfx1250) — matrix operand order repaired and checked by reading, not compiled, not run on hardware
//
#include <hip/hip_runtime.h>
#include <hip/hip_bf16.h>
#include <math.h>

#define D_MODEL 1024
#define NUM_HEADS 16
#define HEAD_DIM 64
#define DFF 4096
#define SEQ 2048
#define BATCH 2
#define ROWS (BATCH * SEQ) /* 4096 */
#define LN_EPS 1e-6f

typedef __attribute__((ext_vector_type(16))) __bf16 v16bf;
typedef __attribute__((ext_vector_type(8)))  float  v8f;

static __device__ __forceinline__ v8f wmma_bf16(v16bf a, v16bf b, v8f c) {
    // (neg_a, A, neg_b, B, c_mod, C, reuse_a, reuse_b)
    return __builtin_amdgcn_wmma_f32_16x16x32_bf16(false, a, false, b, (short)0, c,
                                                   false, false);
}

// build a v16bf fragment from two 16-byte-aligned chunks (b128 loads, LDS or global)
static __device__ __forceinline__ v16bf frag16(const __bf16* p0, const __bf16* p1) {
    union { v16bf v; uint4 q[2]; } u;
    u.q[0] = *reinterpret_cast<const uint4*>(p0);
    u.q[1] = *reinterpret_cast<const uint4*>(p1);
    return u.v;
}

// CDNA5 async copy: 16 bytes per lane, global -> LDS, tracked by ASYNCcnt
static __device__ __forceinline__ void async_cp16(const __bf16* g, __bf16* l) {
    const unsigned lds_off = (unsigned)(size_t)l;       // low 32 bits = LDS byte addr
    const unsigned long long ga = (unsigned long long)(size_t)g;
    asm volatile("global_load_async_to_lds_b128 %0, %1, off"
                 :: "v"(lds_off), "v"(ga) : "memory");
}
static __device__ __forceinline__ void async_wait0() {
    asm volatile("s_wait_asynccnt 0" ::: "memory");
}

// -------------------------------------------------------------------------
// one-time operand conversion kernels
// -------------------------------------------------------------------------
__global__ __launch_bounds__(256)
void cvt_bf16_rm(const float* __restrict__ X, __bf16* __restrict__ Y) {
    const int i = (blockIdx.x * 256 + threadIdx.x) * 4;
    const float4 f = *reinterpret_cast<const float4*>(X + i);
    Y[i + 0] = (__bf16)f.x; Y[i + 1] = (__bf16)f.y;
    Y[i + 2] = (__bf16)f.z; Y[i + 3] = (__bf16)f.w;
}

// Y[n][k] = (bf16) X[k][n]   (X is K x N row-major)
__global__ __launch_bounds__(256)
void cvt_bf16_tr(const float* __restrict__ X, __bf16* __restrict__ Y, int K, int N) {
    const int i = blockIdx.x * 256 + threadIdx.x;   // over N*K
    const int k = i % K, n = i / K;
    Y[(size_t)n * K + k] = (__bf16)X[(size_t)k * N + n];
}

// -------------------------------------------------------------------------
// GEMM: C[M,N] = act(A[M,K] @ Bt[N,K]^T + bias + residual)
// A, Bt bf16; double-buffered async LDS staging (one barrier / K-step, DMA
// overlapped with WMMA); wave tile 32x64 (2x4 WMMA); block tile 64x256.
// OUT_MODE: 0 = f32 row-major, 1 = bf16 row-major, 2 = bf16 V-transposed
// -------------------------------------------------------------------------
template <int OUT_MODE, bool HAS_RES, bool RELU>
__global__ __launch_bounds__(256)
void gemm_bf16_wmma(const __bf16* __restrict__ A, const __bf16* __restrict__ Bt,
                    const float* __restrict__ bias, const float* __restrict__ residual,
                    void* __restrict__ Cout, int M, int N, int K)
{
    __shared__ __attribute__((aligned(16))) __bf16 As[2][64][32];    // [m][k]   8 KB
    __shared__ __attribute__((aligned(16))) __bf16 Bs[2][256][32];   // [n][k]  32 KB

    const int tid  = threadIdx.x;
    const int lane = tid & 31;
    const int wid  = tid >> 5;
    const int half = lane >> 4;
    const int l16  = lane & 15;
    const int wm   = wid & 1;          // 2 waves along M (32 rows each)
    const int wn   = wid >> 1;         // 4 waves along N (64 cols each)
    const int rowBlk = blockIdx.y * 64;
    const int colBlk = blockIdx.x * 256;

    const int ar = tid >> 2, ac = (tid & 3) * 8;   // this thread's A chunk
    auto issue_tile = [&](int k0, int buf) {
        async_cp16(A + (size_t)(rowBlk + ar) * K + k0 + ac, &As[buf][ar][ac]);
        #pragma unroll
        for (int j = 0; j < 4; ++j) {
            const int i = tid + j * 256;
            const int n = i >> 2, c = (i & 3) * 8;
            async_cp16(Bt + (size_t)(colBlk + n) * K + k0 + c, &Bs[buf][n][c]);
        }
    };

    v8f acc[2][4] = {};

    issue_tile(0, 0);                   // prologue: tile 0 -> buffer 0
    int cur = 0;
    for (int k0 = 0; k0 < K; k0 += 32) {
        async_wait0();                  // own copies for buf[cur] complete
        __syncthreads();                // all waves' copies visible; prev reads done
        if (k0 + 32 < K)
            issue_tile(k0 + 32, cur ^ 1);   // DMA next tile while we compute

        // ---- fragments from LDS (ds_load_b128) ----
        v16bf af[2], bfr[4];
        #pragma unroll
        for (int mi = 0; mi < 2; ++mi) {
            const int m = wm * 32 + mi * 16 + l16;
            af[mi] = frag16(&As[cur][m][half * 8], &As[cur][m][16 + half * 8]);
        }
        #pragma unroll
        for (int ni = 0; ni < 4; ++ni) {
            const int n = wn * 64 + ni * 16 + l16;
            bfr[ni] = frag16(&Bs[cur][n][half * 8], &Bs[cur][n][16 + half * 8]);
        }
        #pragma unroll
        for (int mi = 0; mi < 2; ++mi)
            #pragma unroll
            for (int ni = 0; ni < 4; ++ni)
                acc[mi][ni] = wmma_bf16(af[mi], bfr[ni], acc[mi][ni]);
        cur ^= 1;
    }

    // ---- epilogue: slot r is row (r + 8*half), column l16 ----
    #pragma unroll
    for (int mi = 0; mi < 2; ++mi) {
        #pragma unroll
        for (int ni = 0; ni < 4; ++ni) {
            const int col = colBlk + wn * 64 + ni * 16 + l16;
            const float bs = bias[col];
            #pragma unroll
            for (int r = 0; r < 8; ++r) {
                const int row = rowBlk + wm * 32 + mi * 16 + r + 8 * half;
                float v = acc[mi][ni][r] + bs;
                if (HAS_RES) v += residual[(size_t)row * N + col];
                if (RELU)    v = fmaxf(v, 0.0f);
                if (OUT_MODE == 0) {
                    ((float*)Cout)[(size_t)row * N + col] = v;
                } else if (OUT_MODE == 1) {
                    ((__bf16*)Cout)[(size_t)row * N + col] = (__bf16)v;
                } else {
                    // V transposed for attention: Vt[(b*H + h)*Dh + d][s]
                    const int b = row >> 11, s = row & (SEQ - 1);
                    const int h = col >> 6,  d = col & (HEAD_DIM - 1);
                    ((__bf16*)Cout)[((size_t)(b * NUM_HEADS + h) * HEAD_DIM + d) * SEQ + s]
                        = (__bf16)v;
                }
            }
        }
    }
}

// -------------------------------------------------------------------------
// Flash attention (all operands bf16): one wave owns 16 query rows of one
// (b,h); 32 keys/step. Q,K row-major bf16; V transposed [(b,h,d)][s] bf16 so
// every fragment is a pair of b128 loads. Output attn written bf16.
// -------------------------------------------------------------------------
__global__ __launch_bounds__(256)
void flash_attn_wmma(const __bf16* __restrict__ Q, const __bf16* __restrict__ Kg,
                     const __bf16* __restrict__ Vt, __bf16* __restrict__ O)
{
    __shared__ __attribute__((aligned(16))) __bf16 pbuf[8][16 * 32];  // 8 KB

    const int lane = threadIdx.x & 31;
    const int wid  = threadIdx.x >> 5;
    const int half = lane >> 4;
    const int l16  = lane & 15;

    const int w  = blockIdx.x * 8 + wid;
    const int qt = w & (SEQ / 16 - 1);
    const int h  = (w >> 7) & (NUM_HEADS - 1);
    const int b  = w >> 11;

    const size_t base  = (size_t)b * SEQ * D_MODEL + (size_t)h * HEAD_DIM;
    const __bf16* vtb  = Vt + (size_t)(b * NUM_HEADS + h) * HEAD_DIM * SEQ;

    // Q fragments: 16 rows x 64, two K=32 chunks, A-layout (b128 loads)
    v16bf qa[2];
    {
        const __bf16* qp = Q + base + (size_t)(qt * 16 + l16) * D_MODEL;
        #pragma unroll
        for (int c = 0; c < 2; ++c)
            qa[c] = frag16(qp + c * 32 + half * 8, qp + c * 32 + 16 + half * 8);
    }

    v8f acc[4] = {};
    float mrow[8], lrow[8];
    #pragma unroll
    for (int r = 0; r < 8; ++r) { mrow[r] = -3.0e38f; lrow[r] = 0.0f; }

    const float scale = 0.125f;   // 1/sqrt(64)

    for (int t0 = 0; t0 < SEQ; t0 += 32) {
        // scores for two 16-key subtiles
        v8f s[2];
        #pragma unroll
        for (int st = 0; st < 2; ++st) {
            v8f cc = {};
            const __bf16* kp = Kg + base + (size_t)(t0 + st * 16 + l16) * D_MODEL;
            #pragma unroll
            for (int c = 0; c < 2; ++c) {
                // B-layout: lane = key column, K(d) = c*32 + half*8 + e, +16 for e>=8 (table)
                const v16bf kb = frag16(kp + c * 32 + half * 8,
                                        kp + c * 32 + 16 + half * 8);
                cc = wmma_bf16(qa[c], kb, cc);
            }
            s[st] = cc;
        }

        // online softmax per row-slot; write P (bf16) into per-wave LDS tile
        float alpha[8];
        #pragma unroll
        for (int r = 0; r < 8; ++r) {
            float s0 = s[0][r] * scale;
            float s1 = s[1][r] * scale;
            float mx = fmaxf(s0, s1);
            #pragma unroll
            for (int m = 1; m < 16; m <<= 1) mx = fmaxf(mx, __shfl_xor(mx, m, 32));
            const float mnew = fmaxf(mrow[r], mx);
            const float al   = __expf(mrow[r] - mnew);
            const float p0   = __expf(s0 - mnew);
            const float p1   = __expf(s1 - mnew);
            lrow[r] = lrow[r] * al + p0 + p1;
            mrow[r] = mnew;
            alpha[r] = al;
            const int row = r + 8 * half;
            pbuf[wid][row * 32 + l16]      = (__bf16)p0;
            pbuf[wid][row * 32 + 16 + l16] = (__bf16)p1;
        }

        // rescale running output accumulators
        #pragma unroll
        for (int nd = 0; nd < 4; ++nd)
            #pragma unroll
            for (int r = 0; r < 8; ++r) acc[nd][r] *= alpha[r];

        asm volatile("s_wait_dscnt 0" ::: "memory");

        // read P back as a 16x32 A-fragment (ds_load_b128 x2)
        const v16bf pa = frag16(&pbuf[wid][l16 * 32 + half * 8],
                                &pbuf[wid][l16 * 32 + 16 + half * 8]);

        // P @ V across the 4 head-dim column tiles (contiguous b128 loads from Vt)
        #pragma unroll
        for (int nd = 0; nd < 4; ++nd) {
            const __bf16* vp = vtb + (size_t)(nd * 16 + l16) * SEQ + t0 + half * 8;
            const v16bf vb = frag16(vp, vp + 16);
            acc[nd] = wmma_bf16(pa, vb, acc[nd]);
        }
    }

    // finalize: divide by row sums, store bf16
    float inv[8];
    #pragma unroll
    for (int r = 0; r < 8; ++r) {
        float l = lrow[r];
        #pragma unroll
        for (int m = 1; m < 16; m <<= 1) l += __shfl_xor(l, m, 32);
        inv[r] = 1.0f / l;
    }
    #pragma unroll
    for (int nd = 0; nd < 4; ++nd) {
        #pragma unroll
        for (int r = 0; r < 8; ++r) {
            const int row = qt * 16 + r + 8 * half;
            O[base + (size_t)row * D_MODEL + nd * 16 + l16] = (__bf16)(acc[nd][r] * inv[r]);
        }
    }
}

// -------------------------------------------------------------------------
// LayerNorm: one block per row of 1024; optionally also emits bf16 copy
// -------------------------------------------------------------------------
template <bool WRITE_BF>
__global__ __launch_bounds__(256)
void layernorm_kernel(const float* __restrict__ X, const float* __restrict__ gamma,
                      const float* __restrict__ beta, float* __restrict__ Y,
                      __bf16* __restrict__ Ybf)
{
    __shared__ float red[256];
    const int row = blockIdx.x;
    const int tid = threadIdx.x;
    const float* x = X + (size_t)row * D_MODEL;

    float v[4];
    float lsum = 0.0f;
    #pragma unroll
    for (int i = 0; i < 4; ++i) { v[i] = x[tid + i * 256]; lsum += v[i]; }

    red[tid] = lsum; __syncthreads();
    for (int s = 128; s > 0; s >>= 1) {
        if (tid < s) red[tid] += red[tid + s];
        __syncthreads();
    }
    const float mu = red[0] * (1.0f / D_MODEL);
    __syncthreads();

    float lsq = 0.0f;
    #pragma unroll
    for (int i = 0; i < 4; ++i) { const float d = v[i] - mu; lsq += d * d; }
    red[tid] = lsq; __syncthreads();
    for (int s = 128; s > 0; s >>= 1) {
        if (tid < s) red[tid] += red[tid + s];
        __syncthreads();
    }
    const float rs = rsqrtf(red[0] * (1.0f / D_MODEL) + LN_EPS);

    #pragma unroll
    for (int i = 0; i < 4; ++i) {
        const int c = tid + i * 256;
        const float o = (v[i] - mu) * rs * gamma[c] + beta[c];
        Y[(size_t)row * D_MODEL + c] = o;
        if (WRITE_BF) Ybf[(size_t)row * D_MODEL + c] = (__bf16)o;
    }
}

// -------------------------------------------------------------------------
extern "C" void kernel_launch(void* const* d_in, const int* in_sizes, int n_in,
                              void* d_out, int out_size, void* d_ws, size_t ws_size,
                              hipStream_t stream)
{
    (void)in_sizes; (void)n_in; (void)out_size; (void)ws_size;

    const float* x   = (const float*)d_in[0];
    const float* Wq  = (const float*)d_in[1];
    const float* bq  = (const float*)d_in[2];
    const float* Wk  = (const float*)d_in[3];
    const float* bk  = (const float*)d_in[4];
    const float* Wv  = (const float*)d_in[5];
    const float* bv  = (const float*)d_in[6];
    const float* Wo  = (const float*)d_in[7];
    const float* bo  = (const float*)d_in[8];
    const float* W1  = (const float*)d_in[9];
    const float* b1  = (const float*)d_in[10];
    const float* W2  = (const float*)d_in[11];
    const float* b2  = (const float*)d_in[12];
    const float* g1  = (const float*)d_in[13];
    const float* be1 = (const float*)d_in[14];
    const float* g2  = (const float*)d_in[15];
    const float* be2 = (const float*)d_in[16];

    // ---- workspace carve-up (bytes) ----
    char* p = (char*)d_ws;
    const size_t NTOK = (size_t)ROWS * D_MODEL;        // 4M elements
    __bf16* xbf   = (__bf16*)p; p += NTOK * 2;                       //  8 MB
    __bf16* WqT   = (__bf16*)p; p += (size_t)D_MODEL * D_MODEL * 2;  //  2 MB
    __bf16* WkT   = (__bf16*)p; p += (size_t)D_MODEL * D_MODEL * 2;
    __bf16* WvT   = (__bf16*)p; p += (size_t)D_MODEL * D_MODEL * 2;
    __bf16* WoT   = (__bf16*)p; p += (size_t)D_MODEL * D_MODEL * 2;
    __bf16* W1T   = (__bf16*)p; p += (size_t)DFF * D_MODEL * 2;      //  8 MB
    __bf16* W2T   = (__bf16*)p; p += (size_t)D_MODEL * DFF * 2;      //  8 MB
    __bf16* qbf   = (__bf16*)p; p += NTOK * 2;
    __bf16* kbf   = (__bf16*)p; p += NTOK * 2;
    __bf16* vbfT  = (__bf16*)p; p += NTOK * 2;
    __bf16* attnb = (__bf16*)p; p += NTOK * 2;
    __bf16* x1bf  = (__bf16*)p; p += NTOK * 2;
    __bf16* hidbf = (__bf16*)p; p += (size_t)ROWS * DFF * 2;         // 32 MB
    float*  y1    = (float*)p;  p += NTOK * 4;                       // 16 MB
    float*  x1f   = (float*)p;  p += NTOK * 4;
    float*  y2    = (float*)p;  p += NTOK * 4;

    const dim3 blk(256);

    // ---- operand conversion (bf16 everywhere) ----
    cvt_bf16_rm<<<dim3(NTOK / (256 * 4)), blk, 0, stream>>>(x, xbf);
    cvt_bf16_tr<<<dim3((D_MODEL * D_MODEL) / 256), blk, 0, stream>>>(Wq, WqT, D_MODEL, D_MODEL);
    cvt_bf16_tr<<<dim3((D_MODEL * D_MODEL) / 256), blk, 0, stream>>>(Wk, WkT, D_MODEL, D_MODEL);
    cvt_bf16_tr<<<dim3((D_MODEL * D_MODEL) / 256), blk, 0, stream>>>(Wv, WvT, D_MODEL, D_MODEL);
    cvt_bf16_tr<<<dim3((D_MODEL * D_MODEL) / 256), blk, 0, stream>>>(Wo, WoT, D_MODEL, D_MODEL);
    cvt_bf16_tr<<<dim3((D_MODEL * DFF) / 256), blk, 0, stream>>>(W1, W1T, D_MODEL, DFF);
    cvt_bf16_tr<<<dim3((DFF * D_MODEL) / 256), blk, 0, stream>>>(W2, W2T, DFF, D_MODEL);

    const dim3 gProj(D_MODEL / 256, ROWS / 64);        // (4, 64)
    const dim3 gFfn1(DFF / 256, ROWS / 64);            // (16, 64)
    const dim3 gAttn(BATCH * NUM_HEADS * (SEQ / 16) / 8);  // 512 blocks

    // QKV projections (bf16 out; V transposed for attention)
    gemm_bf16_wmma<1, false, false><<<gProj, blk, 0, stream>>>(xbf, WqT, bq, nullptr, qbf,  ROWS, D_MODEL, D_MODEL);
    gemm_bf16_wmma<1, false, false><<<gProj, blk, 0, stream>>>(xbf, WkT, bk, nullptr, kbf,  ROWS, D_MODEL, D_MODEL);
    gemm_bf16_wmma<2, false, false><<<gProj, blk, 0, stream>>>(xbf, WvT, bv, nullptr, vbfT, ROWS, D_MODEL, D_MODEL);
    // attention
    flash_attn_wmma<<<gAttn, blk, 0, stream>>>(qbf, kbf, vbfT, attnb);
    // output projection + residual (f32 out for LN)
    gemm_bf16_wmma<0, true, false><<<gProj, blk, 0, stream>>>(attnb, WoT, bo, x, y1, ROWS, D_MODEL, D_MODEL);
    layernorm_kernel<true><<<dim3(ROWS), blk, 0, stream>>>(y1, g1, be1, x1f, x1bf);
    // FFN
    gemm_bf16_wmma<1, false, true><<<gFfn1, blk, 0, stream>>>(x1bf, W1T, b1, nullptr, hidbf, ROWS, DFF, D_MODEL);
    gemm_bf16_wmma<0, true, false><<<gProj, blk, 0, stream>>>(hidbf, W2T, b2, x1f, y2, ROWS, D_MODEL, DFF);
    layernorm_kernel<false><<<dim3(ROWS), blk, 0, stream>>>(y2, g2, be2, (float*)d_out, nullptr);
}
